// MaskedSelfAttention_44598940401794
// MI455X (gfx1250) — matrix operand order repaired and checked by reading, not compiled, not run on hardware
//
#include <hip/hip_runtime.h>
#include <hip/hip_bf16.h>
#include <stdint.h>

typedef __bf16 bf16t;
typedef __attribute__((ext_vector_type(16))) __bf16 v16bf;
typedef __attribute__((ext_vector_type(8)))  float  v8f;

#define B_  4
#define T_  2048
#define D_  1024
#define H_  16
#define HD_ 64

static __device__ __forceinline__ bf16t f2bf(float f) { return (bf16t)f; }

// ---------------------------------------------------------------------------
// CDNA5 async copy: global -> LDS, 8 bytes per lane, tracked by ASYNCcnt.
// GV addressing form (64-bit VGPR address, SADDR=off).
static __device__ __forceinline__ void async_copy_b64(uint32_t lds_byte,
                                                      const void* gaddr) {
  asm volatile("global_load_async_to_lds_b64 %0, %1, off"
               :: "v"(lds_byte), "v"(gaddr)
               : "memory");
}
static __device__ __forceinline__ void wait_async0() {
  asm volatile("s_wait_asynccnt 0x0" ::: "memory");
}

// ---------------------------------------------------------------------------
// Fragment loaders per CDNA5 WMMA VGPR layouts (ISA 7.12.2), wave32.
// A-fragment 16x32 (16-bit): lane L holds row L&15; pair p holds
// K = (p<4 ? 2p : 16+2(p-4)) + (L>=16 ? 8 : 0).
static __device__ __forceinline__ v16bf load_fragA(const bf16t* p0, int ld) {
  const int lane = threadIdx.x & 31;
  const bf16t* r = p0 + (size_t)(lane & 15) * ld + ((lane >> 4) << 3);
  v16bf a;
#pragma unroll
  for (int p = 0; p < 8; ++p) {
    const int kb = (p < 4) ? (2 * p) : (16 + 2 * (p - 4));
    a[2 * p]     = r[kb];
    a[2 * p + 1] = r[kb + 1];
  }
  return a;
}

// B-fragment 32x16 (16-bit), source laid out column-major p0[n*ld + k]:
// lane L holds column L&15; pair p holds K = (p<4 ? 2p : 16+2(p-4)) + (L>=16 ? 8 : 0).
static __device__ __forceinline__ v16bf load_fragB(const bf16t* p0, int ld) {
  const int lane = threadIdx.x & 31;
  const bf16t* r = p0 + (size_t)(lane & 15) * ld + ((lane >> 4) << 3);
  v16bf b;
#pragma unroll
  for (int v = 0; v < 8; ++v) {
    const int kb = (v < 4) ? (2 * v) : (16 + 2 * (v - 4));
    b[2 * v]     = r[kb];
    b[2 * v + 1] = r[kb + 1];
  }
  return b;
}

// ---------------------------------------------------------------------------
__global__ void convert_f32_bf16(const float* __restrict__ src,
                                 bf16t* __restrict__ dst, int n) {
  int i = blockIdx.x * blockDim.x + threadIdx.x;
  const int stride = gridDim.x * blockDim.x;
  for (; i < n; i += stride) dst[i] = f2bf(src[i]);
}

// ---------------------------------------------------------------------------
// QKV GEMM: [8192,1024] x [1024,3072] + bias, scatter into per-head Q, K and
// transposed V (VT[bh][d][t]) in bf16. Block tile 256(M) x 64(N), 8 waves,
// each wave computes 32x64 (two A-frags share four B-frags -> 8 WMMA/chunk).
// Double-buffered: async-stage chunk k+1 while WMMAs consume chunk k.
__global__ __launch_bounds__(256) void qkv_gemm_kernel(
    const bf16t* __restrict__ Xbf, const bf16t* __restrict__ Wbf,
    const float* __restrict__ bias,
    bf16t* __restrict__ Qws, bf16t* __restrict__ Kws, bf16t* __restrict__ VTws) {
  __shared__ bf16t Asl[2][256 * 40];   // [m][k], k-padded (80 B rows)
  __shared__ bf16t Bsl[2][64 * 40];    // [n][k], k-padded
  const int tid = threadIdx.x, wave = tid >> 5, lane = tid & 31;
  const int n0 = blockIdx.x * 64;
  const int m0 = blockIdx.y * 256;
  const uint32_t AslB = (uint32_t)(uintptr_t)&Asl[0][0];
  v8f acc[2][4] = {};

  auto stageA = [&](int kc, int buf) {
    const uint32_t base = AslB + (uint32_t)buf * (256 * 40 * 2);
#pragma unroll
    for (int it = 0; it < 8; ++it) {
      const int w = tid + it * 256;           // 0..2047 b64 units
      const int m = w >> 3, seg = w & 7;
      async_copy_b64(base + m * 80 + seg * 8,
                     (const char*)Xbf + ((size_t)(m0 + m) * D_ + kc) * 2 + seg * 8);
    }
  };
  auto stageB = [&](int kc, int buf) {
    bf16t* bs = &Bsl[buf][0];
#pragma unroll
    for (int it = 0; it < 8; ++it) {
      const int w = tid + it * 256;
      const int k = w >> 6, n = w & 63;
      bs[n * 40 + k] = Wbf[(size_t)(kc + k) * (3 * D_) + n0 + n];
    }
  };

  stageA(0, 0);
  stageB(0, 0);
  wait_async0();
  __syncthreads();

  int buf = 0;
  for (int kc = 0; kc < D_; kc += 32, buf ^= 1) {
    if (kc + 32 < D_) {                       // prefetch next chunk into buf^1
      stageA(kc + 32, buf ^ 1);
      stageB(kc + 32, buf ^ 1);
    }
    const v16bf a0 = load_fragA(&Asl[buf][(wave * 32) * 40], 40);
    const v16bf a1 = load_fragA(&Asl[buf][(wave * 32 + 16) * 40], 40);
#pragma unroll
    for (int f = 0; f < 4; ++f) {
      const v16bf b = load_fragB(&Bsl[buf][(f * 16) * 40], 40);
      acc[0][f] = __builtin_amdgcn_wmma_f32_16x16x32_bf16(false, a0, false, b,
                                                          (short)0, acc[0][f], false, false);
      acc[1][f] = __builtin_amdgcn_wmma_f32_16x16x32_bf16(false, a1, false, b,
                                                          (short)0, acc[1][f], false, false);
    }
    wait_async0();
    __syncthreads();
  }

  const int colL = lane & 15, rowoff = (lane >> 4) << 3;
#pragma unroll
  for (int sub = 0; sub < 2; ++sub) {
#pragma unroll
    for (int f = 0; f < 4; ++f) {
      const int n = n0 + f * 16 + colL;
      const float bv = bias[n];
#pragma unroll
      for (int j = 0; j < 8; ++j) {
        const int m = m0 + wave * 32 + sub * 16 + j + rowoff;
        const int b = m >> 11, t = m & (T_ - 1);
        const float v = acc[sub][f][j] + bv;
        if (n < D_) {
          const int h = n >> 6, d = n & 63;
          Qws[((size_t)(b * H_ + h) * T_ + t) * HD_ + d] = f2bf(v);
        } else if (n < 2 * D_) {
          const int nn = n - D_, h = nn >> 6, d = nn & 63;
          Kws[((size_t)(b * H_ + h) * T_ + t) * HD_ + d] = f2bf(v);
        } else {
          const int nn = n - 2 * D_, h = nn >> 6, d = nn & 63;
          VTws[((size_t)(b * H_ + h) * HD_ + d) * T_ + t] = f2bf(v);
        }
      }
    }
  }
}

// ---------------------------------------------------------------------------
// Causal flash attention: block = 4 waves = 64 query rows of one (b,h).
// K/V tiles double-buffered via async-to-LDS (overlaps the softmax VALU
// burst); S and softmax stats in registers; P goes through per-wave LDS
// scratch (C-layout -> A-layout) for the PV WMMA.
__global__ __launch_bounds__(128) void attn_kernel(
    const bf16t* __restrict__ Qws, const bf16t* __restrict__ Kws,
    const bf16t* __restrict__ VTws, bf16t* __restrict__ Attn) {
  __shared__ bf16t Ktile[2][32 * 72];       // [key][hd], 144 B rows
  __shared__ bf16t Vtile[2][64 * 40];       // [d][key], 80 B rows
  __shared__ bf16t Ptile[4][16 * 34];       // per-wave P scratch [row][key]
  const int tid = threadIdx.x, wave = tid >> 5, lane = tid & 31;
  const int q0 = blockIdx.x * 64;
  const int bh = blockIdx.y;
  const int qbase = q0 + wave * 16;
  const int colL = lane & 15, rowoff = (lane >> 4) << 3;
  const uint32_t KtB = (uint32_t)(uintptr_t)&Ktile[0][0];
  const uint32_t VtB = (uint32_t)(uintptr_t)&Vtile[0][0];

  auto stageKV = [&](int kpos, int bufi) {
    const uint32_t kbase = KtB + (uint32_t)bufi * (32 * 72 * 2);
    const uint32_t vbase = VtB + (uint32_t)bufi * (64 * 40 * 2);
#pragma unroll
    for (int it = 0; it < 4; ++it) {        // K chunk 32x64 bf16: 512 b64 units
      const int w = tid + it * 128;
      const int key = w >> 4, seg = w & 15;
      async_copy_b64(kbase + key * 144 + seg * 8,
                     (const char*)Kws + ((size_t)bh * T_ + kpos + key) * (HD_ * 2) + seg * 8);
    }
#pragma unroll
    for (int it = 0; it < 4; ++it) {        // V^T chunk 64x32 bf16: 512 b64 units
      const int w = tid + it * 128;
      const int d = w >> 3, seg = w & 7;
      async_copy_b64(vbase + d * 80 + seg * 8,
                     (const char*)VTws + (((size_t)bh * HD_ + d) * T_ + kpos) * 2 + seg * 8);
    }
  };

  const bf16t* Qp = Qws + ((size_t)bh * T_ + qbase) * HD_;
  const v16bf qa0 = load_fragA(Qp + 0, HD_);
  const v16bf qa1 = load_fragA(Qp + 32, HD_);

  float mst[8], lst[8];
  v8f o[4] = {};
#pragma unroll
  for (int j = 0; j < 8; ++j) { mst[j] = -1e30f; lst[j] = 0.0f; }

  const int nkb = (q0 >> 5) + 2;            // causal: keys <= q0+63
  stageKV(0, 0);
  wait_async0();
  __syncthreads();

  int buf = 0;
  for (int kb = 0; kb < nkb; ++kb, buf ^= 1) {
    const int kpos = kb * 32;
    if (kb + 1 < nkb) stageKV(kpos + 32, buf ^ 1);
    const bf16t* Kt = &Ktile[buf][0];
    const bf16t* Vt = &Vtile[buf][0];

    // S = Q * K^T for two 16-key column blocks (accumulate over hd=64)
    v8f s0 = {}, s1 = {};
    {
      const v16bf b00 = load_fragB(Kt, 72);
      const v16bf b01 = load_fragB(Kt + 32, 72);
      s0 = __builtin_amdgcn_wmma_f32_16x16x32_bf16(false, qa0, false, b00, (short)0, s0, false, false);
      s0 = __builtin_amdgcn_wmma_f32_16x16x32_bf16(false, qa1, false, b01, (short)0, s0, false, false);
      const v16bf b10 = load_fragB(Kt + 16 * 72, 72);
      const v16bf b11 = load_fragB(Kt + 16 * 72 + 32, 72);
      s1 = __builtin_amdgcn_wmma_f32_16x16x32_bf16(false, qa0, false, b10, (short)0, s1, false, false);
      s1 = __builtin_amdgcn_wmma_f32_16x16x32_bf16(false, qa1, false, b11, (short)0, s1, false, false);
    }

    const float scale = 0.125f;             // 1/sqrt(64)
    bf16t* Pw = Ptile[wave];
#pragma unroll
    for (int j = 0; j < 8; ++j) {           // online softmax per row
      const int qrow = qbase + j + rowoff;
      float v0 = s0[j] * scale;
      float v1 = s1[j] * scale;
      if (kpos + colL > qrow)      v0 = -1e30f;
      if (kpos + 16 + colL > qrow) v1 = -1e30f;
      float rm = fmaxf(v0, v1);
#pragma unroll
      for (int m = 1; m < 16; m <<= 1) rm = fmaxf(rm, __shfl_xor(rm, m, 32));
      const float mnew  = fmaxf(mst[j], rm);
      const float alpha = __expf(mst[j] - mnew);
      const float p0 = __expf(v0 - mnew);
      const float p1 = __expf(v1 - mnew);
      float rs = p0 + p1;
#pragma unroll
      for (int m = 1; m < 16; m <<= 1) rs += __shfl_xor(rs, m, 32);
      lst[j] = lst[j] * alpha + rs;
      mst[j] = mnew;
#pragma unroll
      for (int f = 0; f < 4; ++f) o[f][j] *= alpha;
      const int prow = j + rowoff;
      Pw[prow * 34 + colL]      = f2bf(p0);
      Pw[prow * 34 + 16 + colL] = f2bf(p1);
    }
    asm volatile("s_wait_dscnt 0x0" ::: "memory");   // wave-local LDS RAW
    const v16bf pa = load_fragA(Pw, 34);
#pragma unroll
    for (int f = 0; f < 4; ++f) {           // O += P * V  (4 x 16 cols of hd)
      const v16bf vb = load_fragB(Vt + (f * 16) * 40, 40);
      o[f] = __builtin_amdgcn_wmma_f32_16x16x32_bf16(false, pa, false, vb, (short)0, o[f], false, false);
    }
    wait_async0();
    __syncthreads();
  }

  const int b = bh >> 4, h = bh & 15;
#pragma unroll
  for (int f = 0; f < 4; ++f) {
    const int d = f * 16 + colL;
#pragma unroll
    for (int j = 0; j < 8; ++j) {
      const int t = qbase + j + rowoff;
      Attn[(size_t)(b * T_ + t) * D_ + h * HD_ + d] = f2bf(o[f][j] / lst[j]);
    }
  }
}

// ---------------------------------------------------------------------------
// Output projection: [8192,1024] x [1024,1024] + bias -> fp32 d_out.
// Same double-buffered 256x64 tile engine as the QKV GEMM.
__global__ __launch_bounds__(256) void out_gemm_kernel(
    const bf16t* __restrict__ Abf, const bf16t* __restrict__ Wbf,
    const float* __restrict__ bias, float* __restrict__ Out) {
  __shared__ bf16t Asl[2][256 * 40];
  __shared__ bf16t Bsl[2][64 * 40];
  const int tid = threadIdx.x, wave = tid >> 5, lane = tid & 31;
  const int n0 = blockIdx.x * 64;
  const int m0 = blockIdx.y * 256;
  const uint32_t AslB = (uint32_t)(uintptr_t)&Asl[0][0];
  v8f acc[2][4] = {};

  auto stageA = [&](int kc, int bufi) {
    const uint32_t base = AslB + (uint32_t)bufi * (256 * 40 * 2);
#pragma unroll
    for (int it = 0; it < 8; ++it) {
      const int w = tid + it * 256;
      const int m = w >> 3, seg = w & 7;
      async_copy_b64(base + m * 80 + seg * 8,
                     (const char*)Abf + ((size_t)(m0 + m) * D_ + kc) * 2 + seg * 8);
    }
  };
  auto stageB = [&](int kc, int bufi) {
    bf16t* bs = &Bsl[bufi][0];
#pragma unroll
    for (int it = 0; it < 8; ++it) {
      const int w = tid + it * 256;
      const int k = w >> 6, n = w & 63;
      bs[n * 40 + k] = Wbf[(size_t)(kc + k) * D_ + n0 + n];
    }
  };

  stageA(0, 0);
  stageB(0, 0);
  wait_async0();
  __syncthreads();

  int buf = 0;
  for (int kc = 0; kc < D_; kc += 32, buf ^= 1) {
    if (kc + 32 < D_) {
      stageA(kc + 32, buf ^ 1);
      stageB(kc + 32, buf ^ 1);
    }
    const v16bf a0 = load_fragA(&Asl[buf][(wave * 32) * 40], 40);
    const v16bf a1 = load_fragA(&Asl[buf][(wave * 32 + 16) * 40], 40);
#pragma unroll
    for (int f = 0; f < 4; ++f) {
      const v16bf b = load_fragB(&Bsl[buf][(f * 16) * 40], 40);
      acc[0][f] = __builtin_amdgcn_wmma_f32_16x16x32_bf16(false, a0, false, b,
                                                          (short)0, acc[0][f], false, false);
      acc[1][f] = __builtin_amdgcn_wmma_f32_16x16x32_bf16(false, a1, false, b,
                                                          (short)0, acc[1][f], false, false);
    }
    wait_async0();
    __syncthreads();
  }

  const int colL = lane & 15, rowoff = (lane >> 4) << 3;
#pragma unroll
  for (int sub = 0; sub < 2; ++sub) {
#pragma unroll
    for (int f = 0; f < 4; ++f) {
      const int n = n0 + f * 16 + colL;
      const float bv = bias[n];
#pragma unroll
      for (int j = 0; j < 8; ++j) {
        const int m = m0 + wave * 32 + sub * 16 + j + rowoff;
        Out[(size_t)m * D_ + n] = acc[sub][f][j] + bv;
      }
    }
  }
}

// ---------------------------------------------------------------------------
extern "C" void kernel_launch(void* const* d_in, const int* in_sizes, int n_in,
                              void* d_out, int out_size, void* d_ws, size_t ws_size,
                              hipStream_t stream) {
  const float* x    = (const float*)d_in[0];
  const float* Wqkv = (const float*)d_in[1];
  const float* bqkv = (const float*)d_in[2];
  const float* Wout = (const float*)d_in[3];
  const float* bout = (const float*)d_in[4];
  float* out = (float*)d_out;

  char* ws = (char*)d_ws;
  size_t off = 0;
  auto alloc = [&](size_t bytes) -> void* {
    void* p = ws + off;
    off += (bytes + 255) & ~(size_t)255;
    return p;
  };
  const size_t NX = (size_t)B_ * T_ * D_;            // 8,388,608
  bf16t* xbf    = (bf16t*)alloc(NX * 2);
  bf16t* wqkvbf = (bf16t*)alloc((size_t)D_ * 3 * D_ * 2);
  bf16t* woutbf = (bf16t*)alloc((size_t)D_ * D_ * 2);
  bf16t* Qws    = (bf16t*)alloc(NX * 2);
  bf16t* Kws    = (bf16t*)alloc(NX * 2);
  bf16t* VTws   = (bf16t*)alloc(NX * 2);
  bf16t* attnbf = (bf16t*)alloc(NX * 2);

  convert_f32_bf16<<<1024, 256, 0, stream>>>(x, xbf, (int)NX);
  convert_f32_bf16<<<1024, 256, 0, stream>>>(Wqkv, wqkvbf, D_ * 3 * D_);
  convert_f32_bf16<<<512, 256, 0, stream>>>(Wout, woutbf, D_ * D_);

  qkv_gemm_kernel<<<dim3(48, 32), 256, 0, stream>>>(xbf, wqkvbf, bqkv, Qws, Kws, VTws);
  attn_kernel<<<dim3(T_ / 64, B_ * H_), 128, 0, stream>>>(Qws, Kws, VTws, attnbf);
  out_gemm_kernel<<<dim3(16, 32), 256, 0, stream>>>(attnbf, woutbf, bout, out);
}
